// RNNDecoder_75642964017648
// MI455X (gfx1250) — matrix operand order repaired and checked by reading, not compiled, not run on hardware
//
#include <hip/hip_runtime.h>

#define VOCAB 50000
#define EMB   256
#define HH    512
#define SS    400
#define BB    64

typedef __attribute__((ext_vector_type(16))) __bf16          v16bf;
typedef __attribute__((ext_vector_type(8)))  float           v8f;
typedef __attribute__((ext_vector_type(8)))  unsigned short  u16x8;
typedef __attribute__((ext_vector_type(16))) unsigned short  u16x16;

__device__ __forceinline__ unsigned short f2bf(float f) {
    unsigned u = __float_as_uint(f);
    unsigned r = u + 0x7FFFu + ((u >> 16) & 1u);   // round-to-nearest-even
    return (unsigned short)(r >> 16);
}

__device__ __forceinline__ v8f wmma_bf16(u16x16 a, u16x16 b, v8f c) {
    return __builtin_amdgcn_wmma_f32_16x16x32_bf16(
        false, __builtin_bit_cast(v16bf, a),
        false, __builtin_bit_cast(v16bf, b),
        (short)0, c, false, false);
}

// ---------------------------------------------------------------- K0: W_h -> W_hT (bf16)
__global__ void k0_wh_transpose(const float* __restrict__ W_h,
                                unsigned short* __restrict__ WhT) {
    int idx = blockIdx.x * 256 + threadIdx.x;        // over 512*1024
    if (idx >= HH * 2 * HH) return;
    int n = idx / (2 * HH);
    int k = idx % (2 * HH);
    WhT[n * (2 * HH) + k] = f2bf(W_h[k * HH + n]);
}

// ---------------------------------------------------------------- K1: LSTM step + x2 + p
__global__ void k1_lstm(const float* __restrict__ emb, const int* __restrict__ inputs,
                        const float* __restrict__ W_ih, const float* __restrict__ b_ih,
                        const float* __restrict__ b_hh, const float* __restrict__ W_s,
                        const float* __restrict__ Wz_w, const float* __restrict__ Wz_b,
                        float* __restrict__ out_h, float* __restrict__ out_c,
                        float* __restrict__ out_p,
                        unsigned short* __restrict__ gs_bf, float* __restrict__ x2_ws) {
    __shared__ float lx[EMB];
    __shared__ float lg[4 * HH];
    __shared__ float lh[HH];
    int b = blockIdx.x, t = threadIdx.x;             // 256 threads
    int tok = inputs[b];
    lx[t] = emb[(size_t)tok * EMB + t];
    __syncthreads();
    for (int j = t; j < 4 * HH; j += 256) {
        float acc = b_ih[j] + b_hh[j];               // h0 == 0 -> no W_hh term
        const float* wr = W_ih + (size_t)j * EMB;
        for (int e = 0; e < EMB; ++e) acc += lx[e] * wr[e];
        lg[j] = acc;
    }
    __syncthreads();
    for (int j = t; j < HH; j += 256) {
        float ig = lg[j], gg = lg[2 * HH + j], og = lg[3 * HH + j];
        float c = (1.f / (1.f + __expf(-ig))) * tanhf(gg);
        float h = (1.f / (1.f + __expf(-og))) * tanhf(c);
        out_c[b * HH + j] = c;
        out_h[b * HH + j] = h;
        gs_bf[b * (3 * HH) + j] = f2bf(h);
        lh[j] = h;
    }
    __syncthreads();
    for (int n = t; n < HH; n += 256) {              // x2 = h @ W_s
        float acc = 0.f;
        for (int k = 0; k < HH; ++k) acc += lh[k] * W_s[k * HH + n];
        x2_ws[b * HH + n] = acc;
    }
    if (t == 0) {                                    // p = sigmoid(h @ Wz_w.T + Wz_b)
        float acc = Wz_b[0];
        for (int k = 0; k < HH; ++k) acc += lh[k] * Wz_w[k];
        out_p[b] = 1.f / (1.f + __expf(-acc));
    }
}

// ---------------------------------------------------------------- K2: fused enc@W_h -> tanh -> scores
__global__ void __launch_bounds__(256) k2_scores(
    const float* __restrict__ enc, const unsigned short* __restrict__ WhT,
    const float* __restrict__ x2, const float* __restrict__ b_attn,
    const float* __restrict__ v_attn, float* __restrict__ scores) {
    __shared__ __align__(16) unsigned short la[16 * 1024];   // 16 rows x 2H, bf16
    __shared__ float lsc[16];
    int tile = blockIdx.x;                 // 0..1599 ; 16 consecutive rows of [S*B]
    int s  = tile >> 2;
    int b0 = (tile & 3) * 16;
    int t = threadIdx.x;

    const float* abase = enc + (size_t)tile * 16 * 1024;
    for (int i = t; i < 16 * 1024; i += 256) la[i] = f2bf(abase[i]);
    if (t < 16) lsc[t] = 0.f;
    __syncthreads();

    int lane = t & 31;
    int wave = t >> 5;                     // 8 waves
    int n = lane & 15;
    int g = lane >> 4;

    for (int ni = 0; ni < 4; ++ni) {
        int nt  = wave + 8 * ni;           // 32 n-tiles cover N = 512
        int ngl = nt * 16 + n;
        const unsigned short* brow = WhT + (size_t)ngl * 1024;
        v8f acc = {0.f, 0.f, 0.f, 0.f, 0.f, 0.f, 0.f, 0.f};
        for (int k0 = 0; k0 < 1024; k0 += 32) {
            u16x8 a0 = *(const u16x8*)&la[n * 1024 + k0 + 8 * g];
            u16x8 a1 = *(const u16x8*)&la[n * 1024 + k0 + 16 + 8 * g];
            u16x8 bl = *(const u16x8*)&brow[k0 + 8 * g];
            u16x8 bh = *(const u16x8*)&brow[k0 + 16 + 8 * g];
            u16x16 av, bv;
            #pragma unroll
            for (int e = 0; e < 8; ++e) {
                av[e] = a0[e]; av[e + 8] = a1[e];
                bv[e] = bl[e]; bv[e + 8] = bh[e];
            }
            acc = wmma_bf16(av, bv, acc);
        }
        float van = v_attn[ngl];
        float ban = b_attn[ngl];
        #pragma unroll
        for (int r = 0; r < 8; ++r) {
            int mrow = 8 * g + r;          // C-tile row held by this lane's VGPR r
            float val = acc[r] + x2[(b0 + mrow) * HH + ngl] + ban;
            float contrib = tanhf(val) * van;
            #pragma unroll
            for (int off = 8; off > 0; off >>= 1)
                contrib += __shfl_xor(contrib, off, 16);
            if (n == 0) atomicAdd(&lsc[mrow], contrib);
        }
    }
    __syncthreads();
    if (t < 16) scores[s * BB + b0 + t] = lsc[t];
}

// ---------------------------------------------------------------- K3: softmax over S (per column b)
__global__ void k3_softmax(const float* __restrict__ scores, float* __restrict__ attn) {
    __shared__ float red[256];
    int b = blockIdx.x, t = threadIdx.x;
    float mx = -1e30f;
    for (int s = t; s < SS; s += 256) mx = fmaxf(mx, scores[s * BB + b]);
    red[t] = mx; __syncthreads();
    for (int o = 128; o > 0; o >>= 1) { if (t < o) red[t] = fmaxf(red[t], red[t + o]); __syncthreads(); }
    mx = red[0]; __syncthreads();
    float sum = 0.f;
    for (int s = t; s < SS; s += 256) sum += __expf(scores[s * BB + b] - mx);
    red[t] = sum; __syncthreads();
    for (int o = 128; o > 0; o >>= 1) { if (t < o) red[t] += red[t + o]; __syncthreads(); }
    float inv = 1.f / red[0];
    for (int s = t; s < SS; s += 256) attn[s * BB + b] = __expf(scores[s * BB + b] - mx) * inv;
}

// ---------------------------------------------------------------- K4: context = attn-weighted enc; fill gs tail
__global__ void k4_context(const float* __restrict__ enc, const float* __restrict__ attn,
                           unsigned short* __restrict__ gs_bf) {
    __shared__ float lat[SS];
    int b = blockIdx.x, t = threadIdx.x;
    for (int s = t; s < SS; s += 256) lat[s] = attn[s * BB + b];
    __syncthreads();
    for (int d = t; d < 2 * HH; d += 256) {
        float acc = 0.f;
        for (int s = 0; s < SS; ++s)
            acc += lat[s] * enc[((size_t)s * BB + b) * (2 * HH) + d];
        gs_bf[b * (3 * HH) + HH + d] = f2bf(acc);
    }
}

// ---------------------------------------------------------------- K5: vocab GEMM + selu + p*  (the HBM stream)
__global__ void __launch_bounds__(128) k5_vocab(
    const unsigned short* __restrict__ gs_bf, const float* __restrict__ V_w,
    const float* __restrict__ V_b, const float* __restrict__ p,
    float* __restrict__ ext) {
    int t = threadIdx.x;
    int lane = t & 31;
    int wave = t >> 5;                     // 4 waves -> m0 = 0,16,32,48
    int m0 = wave * 16;
    int n = lane & 15;
    int g = lane >> 4;
    int ncol = blockIdx.x * 16 + n;        // 3125 * 16 = 50000

    const unsigned short* arow = gs_bf + (size_t)(m0 + n) * 1536;  // A row = lane%16
    const float* brow = V_w + (size_t)ncol * 1536;                 // B col = V_w row
    v8f acc = {0.f, 0.f, 0.f, 0.f, 0.f, 0.f, 0.f, 0.f};
    for (int k0 = 0; k0 < 1536; k0 += 32) {
        __builtin_prefetch(brow + k0 + 256, 0, 1);
        u16x8 a0 = *(const u16x8*)&arow[k0 + 16 * g];
        u16x8 a1 = *(const u16x8*)&arow[k0 + 16 * g + 8];
        const float* bp = brow + k0 + 16 * g;
        float4 f0 = *(const float4*)(bp);
        float4 f1 = *(const float4*)(bp + 4);
        float4 f2 = *(const float4*)(bp + 8);
        float4 f3 = *(const float4*)(bp + 12);
        u16x16 av, bv;
        #pragma unroll
        for (int e = 0; e < 8; ++e) { av[e] = a0[e]; av[e + 8] = a1[e]; }
        bv[0]  = f2bf(f0.x); bv[1]  = f2bf(f0.y); bv[2]  = f2bf(f0.z); bv[3]  = f2bf(f0.w);
        bv[4]  = f2bf(f1.x); bv[5]  = f2bf(f1.y); bv[6]  = f2bf(f1.z); bv[7]  = f2bf(f1.w);
        bv[8]  = f2bf(f2.x); bv[9]  = f2bf(f2.y); bv[10] = f2bf(f2.z); bv[11] = f2bf(f2.w);
        bv[12] = f2bf(f3.x); bv[13] = f2bf(f3.y); bv[14] = f2bf(f3.z); bv[15] = f2bf(f3.w);
        acc = wmma_bf16(av, bv, acc);
    }
    float vb = V_b[ncol];
    #pragma unroll
    for (int r = 0; r < 8; ++r) {
        int bi = m0 + 8 * g + r;
        float x = acc[r] + vb;
        float sv = 1.0507009873554805f *
                   (x > 0.f ? x : 1.6732632423543772f * (__expf(x) - 1.f));
        ext[(size_t)bi * VOCAB + ncol] = p[bi] * sv;
    }
}

// ---------------------------------------------------------------- K6: pointer-copy scatter-add
__global__ void k6_scatter(const int* __restrict__ ewi, const float* __restrict__ attn,
                           const float* __restrict__ p, float* __restrict__ ext) {
    int idx = blockIdx.x * 256 + threadIdx.x;
    if (idx >= SS * BB) return;
    int b = idx % BB;
    int w = ewi[idx];                      // encoder_word_idx[s][b]
    float val = (1.f - p[b]) * attn[idx];  // (1-p) * attn[b][s]
    atomicAdd(&ext[(size_t)b * VOCAB + w], val);
}

// ----------------------------------------------------------------
extern "C" void kernel_launch(void* const* d_in, const int* in_sizes, int n_in,
                              void* d_out, int out_size, void* d_ws, size_t ws_size,
                              hipStream_t stream) {
    const float* enc   = (const float*)d_in[0];
    const float* emb   = (const float*)d_in[1];
    const float* W_ih  = (const float*)d_in[2];
    // d_in[3] = W_hh (unused: h0 == 0)
    const float* b_ih  = (const float*)d_in[4];
    const float* b_hh  = (const float*)d_in[5];
    const float* W_h   = (const float*)d_in[6];
    const float* W_s   = (const float*)d_in[7];
    const float* v_at  = (const float*)d_in[8];
    const float* b_at  = (const float*)d_in[9];
    const float* V_w   = (const float*)d_in[10];
    const float* V_b   = (const float*)d_in[11];
    const float* Wz_w  = (const float*)d_in[12];
    const float* Wz_b  = (const float*)d_in[13];
    const int*   inp   = (const int*)d_in[14];
    const int*   ewi   = (const int*)d_in[15];

    float* out = (float*)d_out;
    float* ext   = out;                                    // [B, V]
    float* out_h = out + (size_t)BB * VOCAB;               // [B, H]
    float* out_c = out_h + (size_t)BB * HH;                // [B, H]
    float* out_p = out_c + (size_t)BB * HH;                // [B]
    float* attnT = out_p + BB;                             // [S, B]

    char* ws = (char*)d_ws;
    unsigned short* WhT   = (unsigned short*)(ws);                    // 512*1024 bf16  (1 MB)
    unsigned short* gs_bf = (unsigned short*)(ws + 1048576);          // 64*1536 bf16   (192 KB)
    float*          x2    = (float*)(ws + 1245184);                   // 64*512 f32     (128 KB)
    float*          sco   = (float*)(ws + 1376256);                   // 400*64 f32     (100 KB)

    k0_wh_transpose<<<2048, 256, 0, stream>>>(W_h, WhT);
    k1_lstm<<<BB, 256, 0, stream>>>(emb, inp, W_ih, b_ih, b_hh, W_s, Wz_w, Wz_b,
                                    out_h, out_c, out_p, gs_bf, x2);
    k2_scores<<<1600, 256, 0, stream>>>(enc, WhT, x2, b_at, v_at, sco);
    k3_softmax<<<BB, 256, 0, stream>>>(sco, attnT);
    k4_context<<<BB, 256, 0, stream>>>(enc, attnT, gs_bf);
    k5_vocab<<<3125, 128, 0, stream>>>(gs_bf, V_w, V_b, out_p, ext);
    k6_scatter<<<100, 256, 0, stream>>>(ewi, attnT, out_p, ext);
}
